// ReservoirBlock_41652592837109
// MI455X (gfx1250) — matrix operand order repaired and checked by reading, not compiled, not run on hardware
//
#include <hip/hip_runtime.h>
#include <math.h>

// ---------------- problem constants ----------------
#define B_   16
#define T_   2048
#define D_   256
#define N_   512
#define TOT_ 1024
#define H_   1024
#define E_   256
#define LR_  0.3f
#define SLOPE_ 0.01f

typedef __bf16 bf16;
typedef __attribute__((ext_vector_type(16))) __bf16 v16bf;
typedef __attribute__((ext_vector_type(8)))  __bf16 v8bf;
typedef __attribute__((ext_vector_type(8)))  float  v8f;
typedef __attribute__((ext_vector_type(4)))  unsigned int u32x4;
typedef __attribute__((ext_vector_type(8)))  int i32x8;
typedef __attribute__((ext_vector_type(4)))  int i32x4;

// ---------------- workspace layout ----------------
constexpr size_t AL(size_t x) { return (x + 255) & ~size_t(255); }
constexpr size_t OFF_SYNC = 0;                                   // 64 u32
constexpr size_t OFF_SBF  = AL(OFF_SYNC + 256);                  // 2*16*512 bf16
constexpr size_t OFF_SF32 = AL(OFF_SBF + size_t(2)*16*512*2);    // 2*16*512 f32
constexpr size_t OFF_XBF  = AL(OFF_SF32 + size_t(2)*16*512*4);   // B*T*D bf16
constexpr size_t OFF_W0   = AL(OFF_XBF + size_t(B_)*T_*D_*2);
constexpr size_t OFF_W1   = AL(OFF_W0 + size_t(N_)*N_*2);
constexpr size_t OFF_WIN0 = AL(OFF_W1 + size_t(N_)*N_*2);
constexpr size_t OFF_WIN1 = AL(OFF_WIN0 + size_t(N_)*D_*2);
constexpr size_t OFF_WR1  = AL(OFF_WIN1 + size_t(N_)*D_*2);
constexpr size_t OFF_WR2  = AL(OFF_WR1 + size_t(H_)*TOT_*2);
constexpr size_t OFF_U0   = AL(OFF_WR2 + size_t(E_)*H_*2);
constexpr size_t OFF_U1   = AL(OFF_U0 + size_t(T_)*B_*N_*4);
constexpr size_t OFF_HIST = AL(OFF_U1 + size_t(T_)*B_*N_*4);
constexpr size_t WS_NEED  = OFF_HIST + size_t(T_)*B_*TOT_*2;

// ---------------- WMMA helpers ----------------
// concat two 16B (8 x bf16) chunks into the 16-element WMMA operand
__device__ __forceinline__ v16bf ld16(const bf16* p0, const bf16* p1) {
  v8bf lo = *(const v8bf*)p0;
  v8bf hi = *(const v8bf*)p1;
  v16bf r;
#pragma unroll
  for (int i = 0; i < 8; ++i) { r[i] = lo[i]; r[i + 8] = hi[i]; }
  return r;
}
// A tile 16x32 (M x K), row-major source with 'stride' elems between rows.
// lanes 0-15: row=lane, K = k..k+7 & k+16..k+23 ; lanes 16-31: same rows, K +8.
__device__ __forceinline__ v16bf load_A(const bf16* base, int stride, int k, int lane) {
  int m = lane & 15;
  int o = (lane >> 4) ? 8 : 0;
  const bf16* p = base + (size_t)m * stride + k + o;
  return ld16(p, p + 16);
}
// B tile 32x16 (K x N) with B[k][n] = W[n][k], W row-major (stride elems/row).
// lanes 0-15: col=lane, K = k..k+7 & k+16..k+23 ; lanes 16-31: col=lane-16, K +8.
__device__ __forceinline__ v16bf load_B(const bf16* w, int stride, int nbase, int k, int lane) {
  int n = nbase + (lane & 15);
  int o = (lane >> 4) ? 8 : 0;
  const bf16* p = w + (size_t)n * stride + k + o;
  return ld16(p, p + 16);
}
__device__ __forceinline__ v8f wmma_bf16(v16bf a, v16bf b, v8f c) {
  return __builtin_amdgcn_wmma_f32_16x16x32_bf16(false, a, false, b, (short)0, c, false, false);
}

// ---------------- kernels ----------------
__global__ void k_zero(unsigned* p, size_t n) {
  for (size_t i = blockIdx.x * (size_t)blockDim.x + threadIdx.x; i < n;
       i += (size_t)gridDim.x * blockDim.x) p[i] = 0u;
}

__global__ void k_cvt(const float* __restrict__ src, bf16* __restrict__ dst, int n) {
  for (int i = blockIdx.x * blockDim.x + threadIdx.x; i < n; i += gridDim.x * blockDim.x)
    dst[i] = (bf16)src[i];
}

// U_r[t][b][n] = sum_d x[b][t][d] * Win_r[n][d]   (one 16x16 output tile per wave)
__global__ void __launch_bounds__(256) k_drive(const bf16* __restrict__ xbf,
                                               const bf16* __restrict__ win0,
                                               const bf16* __restrict__ win1,
                                               float* __restrict__ u0,
                                               float* __restrict__ u1) {
  int lane = threadIdx.x & 31;
  int gw = blockIdx.x * (blockDim.x >> 5) + (threadIdx.x >> 5);  // 131072 waves
  int r = gw >> 16;
  int rem = gw & 65535;
  int t = rem >> 5;
  int nt = rem & 31;
  const bf16* win = r ? win1 : win0;
  float* u = r ? u1 : u0;
  const bf16* abase = xbf + (size_t)t * D_;  // row m at abase + m*T_*D_
  v8f c = {};
#pragma unroll 4
  for (int k = 0; k < D_; k += 32) {
    v16bf a = load_A(abase, T_ * D_, k, lane);
    v16bf b = load_B(win, D_, nt * 16, k, lane);
    c = wmma_bf16(a, b, c);
  }
  int col = nt * 16 + (lane & 15);
  int rb = (lane >> 4) ? 8 : 0;
#pragma unroll
  for (int i = 0; i < 8; ++i)
    u[((size_t)t * B_ + rb + i) * N_ + col] = c[i];
}

// Recurrence: grid = 16 WGs (2 reservoirs x 8 column slices of 64), 256 thr (8 waves).
// Each WG keeps its W slice (64x512 bf16 = 64KB) resident in LDS (loaded by TDM).
__global__ void __launch_bounds__(256) k_recur(const bf16* __restrict__ w0bf,
                                               const bf16* __restrict__ w1bf,
                                               const float* __restrict__ u0,
                                               const float* __restrict__ u1,
                                               bf16* sbf, float* sf32,
                                               bf16* hist, unsigned* syncp) {
  __shared__ __align__(16) bf16 lw[64 * 512];     // 64 KB W slice
  __shared__ float red[4][2][32][8];              // 8 KB partial C

  const int r = blockIdx.x >> 3;
  const int slice = blockIdx.x & 7;
  const int n0 = slice * 64;
  const bf16* wbf = r ? w1bf : w0bf;
  const float* u = r ? u1 : u0;
  bf16* sb = sbf + (size_t)r * 16 * 512;
  float* sf = sf32 + (size_t)r * 16 * 512;
  unsigned* cnt = syncp + r * 2;
  unsigned* gen = syncp + r * 2 + 1;

  // ---- stage W slice into LDS ----
#if defined(__has_builtin) && __has_builtin(__builtin_amdgcn_tensor_load_to_lds)
  if (threadIdx.x < 32) {  // one wave issues the TDM descriptor
    unsigned long long ga = (unsigned long long)(size_t)(wbf + (size_t)n0 * 512);
    unsigned ldsa = (unsigned)(size_t)(void*)&lw[0];   // low 32 bits = LDS offset
    u32x4 g0;
    g0.x = 1u;                                          // count = 1 valid descriptor
    g0.y = ldsa;                                        // lds_addr
    g0.z = (unsigned)(ga & 0xFFFFFFFFu);                // global_addr[31:0]
    g0.w = (unsigned)((ga >> 32) & 0x01FFFFFFu) | 0x80000000u;  // addr[56:32], type=2
    i32x8 g1;
    g1[0] = 0x00010000;            // data_size = 1 (2 bytes)
    g1[1] = (int)(512u << 16);     // tensor_dim0 = 512 (low 16 in [31:16])
    g1[2] = (int)(512u << 16);     // tensor_dim1 = 512
    g1[3] = (int)(512u << 16);     // tile_dim0 = 512
    g1[4] = 64;                    // tile_dim1 = 64 rows
    g1[5] = 512;                   // tensor_dim0_stride = 512 elems
    g1[6] = 0;
    g1[7] = 0;
    i32x4 z4 = {0, 0, 0, 0};
    i32x8 z8 = {0, 0, 0, 0, 0, 0, 0, 0};
    __builtin_amdgcn_tensor_load_to_lds(g0, g1, z4, z4, z8, 0);
    __builtin_amdgcn_s_wait_tensorcnt(0);
  }
#else
  {
    const bf16* src = wbf + (size_t)n0 * 512;
    for (int i = threadIdx.x; i < 64 * 512 / 8; i += blockDim.x)
      ((v8bf*)lw)[i] = ((const v8bf*)src)[i];
  }
#endif
  __syncthreads();

  const int lane = threadIdx.x & 31;
  const int wv = threadIdx.x >> 5;
  const int nt = wv & 3;   // N sub-tile (16 cols each)
  const int q = wv >> 2;   // K half (256 each)

  for (int t = 0; t < T_; ++t) {
    v8f c = {};
#pragma unroll 4
    for (int j = 0; j < 8; ++j) {
      int k = q * 256 + j * 32;
      v16bf a = load_A(sb, 512, k, lane);                 // s_{t-1} (bf16, global/L2)
      v16bf b = load_B((const bf16*)lw, 512, nt * 16, k, lane);  // W slice (LDS)
      c = wmma_bf16(a, b, c);
    }
#pragma unroll
    for (int i = 0; i < 8; ++i) red[nt][q][lane][i] = c[i];
    __syncthreads();

    // 1024 outputs (16 batch x 64 cols), 4 per thread: reduce + leak + tanh + store
#pragma unroll
    for (int e4 = 0; e4 < 4; ++e4) {
      int e = threadIdx.x + e4 * 256;
      int ent = e >> 8, i = (e >> 5) & 7, l = e & 31;
      int row = (l < 16) ? i : i + 8;                 // batch index
      int col = n0 + ent * 16 + (l & 15);             // reservoir column
      float acc = red[ent][0][l][i] + red[ent][1][l][i];
      float uu = u[((size_t)t * B_ + row) * N_ + col];
      float so = sf[row * 512 + col];
      float v = (1.0f - LR_) * so + LR_ * tanhf(uu + acc);
      sf[row * 512 + col] = v;
      sb[row * 512 + col] = (bf16)v;
      hist[((size_t)t * B_ + row) * TOT_ + r * N_ + col] = (bf16)v;
    }
    __threadfence();   // release new state to agent scope
    __syncthreads();

    // ---- step barrier across the 8 WGs of this reservoir ----
    if (threadIdx.x == 0) {
      if (__builtin_amdgcn_cluster_id_x() != 0) {
        __builtin_amdgcn_s_cluster_barrier();        // CDNA5 cluster path
      } else {
        unsigned arr = __hip_atomic_fetch_add(cnt, 1u, __ATOMIC_ACQ_REL,
                                              __HIP_MEMORY_SCOPE_AGENT);
        if (arr == (unsigned)(8 * (t + 1) - 1))
          __hip_atomic_store(gen, (unsigned)(t + 1), __ATOMIC_RELEASE,
                             __HIP_MEMORY_SCOPE_AGENT);
        while (__hip_atomic_load(gen, __ATOMIC_ACQUIRE,
                                 __HIP_MEMORY_SCOPE_AGENT) < (unsigned)(t + 1))
          __builtin_amdgcn_s_sleep(1);
      }
    }
    __syncthreads();
    __threadfence();   // acquire: invalidate WGP$ so sb re-reads see peers' writes
  }
}

// Fused readout: per 16-row tile of hist: h1 = LeakyReLU(hist@Wr1^T+br1) -> LDS(bf16),
// out = h1@Wr2^T + br2, with [t*16+b] -> out[b][t] remap.
__global__ void __launch_bounds__(256) k_readout(const bf16* __restrict__ hist,
                                                 const bf16* __restrict__ wr1,
                                                 const bf16* __restrict__ wr2,
                                                 const float* __restrict__ br1,
                                                 const float* __restrict__ br2,
                                                 float* __restrict__ out) {
  __shared__ __align__(16) bf16 h1[16 * TOT_];   // 32 KB
  const int rt = blockIdx.x;                     // 2048 row tiles
  const int lane = threadIdx.x & 31;
  const int wv = threadIdx.x >> 5;
  const bf16* abase = hist + (size_t)rt * 16 * TOT_;
  const int rb = (lane >> 4) ? 8 : 0;

  // GEMM1: each wave owns 8 of the 64 H-tiles
  for (int j = 0; j < 8; ++j) {
    int ntile = wv * 8 + j;
    int n = ntile * 16 + (lane & 15);
    v8f c = {};
#pragma unroll 4
    for (int k = 0; k < TOT_; k += 32)
      c = wmma_bf16(load_A(abase, TOT_, k, lane), load_B(wr1, TOT_, ntile * 16, k, lane), c);
    float bias = br1[n];
#pragma unroll
    for (int i = 0; i < 8; ++i) {
      float v = c[i] + bias;
      v = (v >= 0.0f) ? v : SLOPE_ * v;
      h1[(rb + i) * TOT_ + n] = (bf16)v;
    }
  }
  __syncthreads();

  // GEMM2: each wave owns 2 of the 16 E-tiles
  for (int j = 0; j < 2; ++j) {
    int ntile = wv * 2 + j;
    int n = ntile * 16 + (lane & 15);
    v8f c = {};
#pragma unroll 4
    for (int k = 0; k < TOT_; k += 32)
      c = wmma_bf16(load_A((const bf16*)h1, TOT_, k, lane),
                    load_B(wr2, TOT_, ntile * 16, k, lane), c);
    float bias = br2[n];
#pragma unroll
    for (int i = 0; i < 8; ++i) {
      int row = rt * 16 + rb + i;     // hist row = t*16 + b
      int t = row >> 4, b = row & 15;
      out[((size_t)b * T_ + t) * E_ + n] = c[i] + bias;
    }
  }
}

// ---------------- host launcher ----------------
extern "C" void kernel_launch(void* const* d_in, const int* in_sizes, int n_in,
                              void* d_out, int out_size, void* d_ws, size_t ws_size,
                              hipStream_t stream) {
  (void)in_sizes; (void)n_in; (void)out_size;
  if (ws_size < WS_NEED) return;   // workspace too small to run safely

  const float* x    = (const float*)d_in[0];
  const float* W0   = (const float*)d_in[1];
  const float* Win0 = (const float*)d_in[2];
  const float* W1   = (const float*)d_in[3];
  const float* Win1 = (const float*)d_in[4];
  const float* Wr1  = (const float*)d_in[5];
  const float* br1  = (const float*)d_in[6];
  const float* Wr2  = (const float*)d_in[7];
  const float* br2  = (const float*)d_in[8];
  float* out = (float*)d_out;

  char* ws = (char*)d_ws;
  unsigned* syncp = (unsigned*)(ws + OFF_SYNC);
  bf16* sbf    = (bf16*)(ws + OFF_SBF);
  float* sf32  = (float*)(ws + OFF_SF32);
  bf16* xbf    = (bf16*)(ws + OFF_XBF);
  bf16* w0bf   = (bf16*)(ws + OFF_W0);
  bf16* w1bf   = (bf16*)(ws + OFF_W1);
  bf16* win0bf = (bf16*)(ws + OFF_WIN0);
  bf16* win1bf = (bf16*)(ws + OFF_WIN1);
  bf16* wr1bf  = (bf16*)(ws + OFF_WR1);
  bf16* wr2bf  = (bf16*)(ws + OFF_WR2);
  float* u0    = (float*)(ws + OFF_U0);
  float* u1    = (float*)(ws + OFF_U1);
  bf16* hist   = (bf16*)(ws + OFF_HIST);

  // 1) zero sync counters + fp32/bf16 state (region [0, OFF_XBF))
  k_zero<<<64, 256, 0, stream>>>((unsigned*)ws, OFF_XBF / 4);

  // 2) fp32 -> bf16 conversions
  k_cvt<<<2048, 256, 0, stream>>>(x,    xbf,    B_ * T_ * D_);
  k_cvt<<<256, 256, 0, stream>>>(W0,   w0bf,   N_ * N_);
  k_cvt<<<256, 256, 0, stream>>>(W1,   w1bf,   N_ * N_);
  k_cvt<<<128, 256, 0, stream>>>(Win0, win0bf, N_ * D_);
  k_cvt<<<128, 256, 0, stream>>>(Win1, win1bf, N_ * D_);
  k_cvt<<<512, 256, 0, stream>>>(Wr1,  wr1bf,  H_ * TOT_);
  k_cvt<<<256, 256, 0, stream>>>(Wr2,  wr2bf,  E_ * H_);

  // 3) input drives (parallel over T): 2 * 2048 * 32 tiles, 8 waves/block
  k_drive<<<16384, 256, 0, stream>>>(xbf, win0bf, win1bf, u0, u1);

  // 4) sequential reservoir scan: 2 reservoirs x 8 persistent WGs
  k_recur<<<16, 256, 0, stream>>>(w0bf, w1bf, u0, u1, sbf, sf32, hist, syncp);

  // 5) fused readout: 2048 row tiles
  k_readout<<<2048, 256, 0, stream>>>(hist, wr1bf, wr2bf, br1, br2, out);
}
